// HungarianMatcherAndLoss_19610820673560
// MI455X (gfx1250) — compile-verified, matrix operand order checked by reading, not run on hardware
//
#include <hip/hip_runtime.h>
#include <hip/hip_bf16.h>
#include <math.h>

#define BS 64
#define NQ 300
#define NC 10
#define NT 30

typedef __attribute__((ext_vector_type(2)))  float    v2f;
typedef __attribute__((ext_vector_type(8)))  float    v8f;
typedef __attribute__((ext_vector_type(16))) _Float16 v16h;

#if __has_builtin(__builtin_amdgcn_wmma_f32_16x16x4_f32)
#define HAVE_WMMA_F32X4 1
#endif

// ---------------------------------------------------------------------------
// helpers
// ---------------------------------------------------------------------------
__device__ inline float giou_pair(float ax1, float ay1, float ax2, float ay2, float areaA,
                                  float bx1, float by1, float bx2, float by2, float areaB)
{
    float ltx = fmaxf(ax1, bx1), lty = fmaxf(ay1, by1);
    float rbx = fminf(ax2, bx2), rby = fminf(ay2, by2);
    float w = fmaxf(rbx - ltx, 0.f), h = fmaxf(rby - lty, 0.f);
    float inter = w * h;
    float uni   = areaA + areaB - inter;
    float iou   = inter / uni;
    float cx1 = fminf(ax1, bx1), cy1 = fminf(ay1, by1);
    float cx2 = fmaxf(ax2, bx2), cy2 = fmaxf(ay2, by2);
    float cw = fmaxf(cx2 - cx1, 0.f), ch = fmaxf(cy2 - cy1, 0.f);
    float areac = cw * ch;
    return iou - (areac - uni) / areac;
}

// ---------------------------------------------------------------------------
// Kernel 1: per-batch cost block in LDS + wave-parallel Jonker-Volgenant
//           (cost.T is 30 rows x 300 cols) + fused matched-pair losses.
// One workgroup (single wave32) per batch.
// ---------------------------------------------------------------------------
__global__ __launch_bounds__(32)
void hungarian_kernel(const float* __restrict__ logits,
                      const float* __restrict__ pboxes,
                      const int*   __restrict__ labels,
                      const float* __restrict__ tboxes,
                      int*   __restrict__ q2t,        // [BS*NQ] matched target idx or -1
                      float* __restrict__ bbox_part,  // [BS]
                      float* __restrict__ giou_part)  // [BS] sum of (1 - giou)
{
    __shared__ float  costT[NT][NQ];      // cost transposed: [target][query]
    __shared__ float  traw[NT][4];        // target boxes cxcywh
    __shared__ float  txy[NT][4];         // target boxes xyxy
    __shared__ float  tarea[NT];
    __shared__ int    tlab[NT];
    __shared__ double u_[NT + 1];
    __shared__ double v_[NQ + 1];
    __shared__ double minv[NQ + 1];
    __shared__ int    p_[NQ + 1];
    __shared__ int    way_[NQ + 1];
    __shared__ unsigned char used_[NQ + 1];

    const int b    = blockIdx.x;
    const int lane = threadIdx.x;
    const double INF = 1e300;

    // ---- stage targets ----
    for (int t = lane; t < NT; t += 32) {
        float cx = tboxes[((b * NT) + t) * 4 + 0];
        float cy = tboxes[((b * NT) + t) * 4 + 1];
        float w  = tboxes[((b * NT) + t) * 4 + 2];
        float h  = tboxes[((b * NT) + t) * 4 + 3];
        traw[t][0] = cx; traw[t][1] = cy; traw[t][2] = w; traw[t][3] = h;
        float x1 = cx - 0.5f * w, y1 = cy - 0.5f * h;
        float x2 = cx + 0.5f * w, y2 = cy + 0.5f * h;
        txy[t][0] = x1; txy[t][1] = y1; txy[t][2] = x2; txy[t][3] = y2;
        tarea[t]  = (x2 - x1) * (y2 - y1);
        tlab[t]   = labels[b * NT + t];
    }
    __syncthreads();

    // ---- fill cost block: C[q][t] = 5*L1 + 2*(-prob[label_t]) + 2*(-giou) ----
    for (int q = lane; q < NQ; q += 32) {
        const float* lg = logits + (size_t)(b * NQ + q) * NC;
        float prob[NC];
        #pragma unroll
        for (int c = 0; c < NC; ++c) prob[c] = 1.f / (1.f + expf(-lg[c]));

        const float* pb = pboxes + (size_t)(b * NQ + q) * 4;
        float cx = pb[0], cy = pb[1], w = pb[2], h = pb[3];
        float qx1 = cx - 0.5f * w, qy1 = cy - 0.5f * h;
        float qx2 = cx + 0.5f * w, qy2 = cy + 0.5f * h;
        float qarea = (qx2 - qx1) * (qy2 - qy1);

        for (int t = 0; t < NT; ++t) {
            float l1 = fabsf(cx - traw[t][0]) + fabsf(cy - traw[t][1]) +
                       fabsf(w  - traw[t][2]) + fabsf(h  - traw[t][3]);
            float g  = giou_pair(qx1, qy1, qx2, qy2, qarea,
                                 txy[t][0], txy[t][1], txy[t][2], txy[t][3], tarea[t]);
            costT[t][q] = 5.0f * l1 - 2.0f * prob[tlab[t]] - 2.0f * g;
        }
    }

    // ---- JV init ----
    for (int j = lane; j <= NQ; j += 32) { v_[j] = 0.0; p_[j] = 0; way_[j] = 0; }
    for (int i = lane; i <= NT; i += 32) u_[i] = 0.0;
    __syncthreads();

    // ---- JV rows ----
    for (int i = 1; i <= NT; ++i) {
        if (lane == 0) p_[0] = i;
        for (int j = lane; j <= NQ; j += 32) { minv[j] = INF; used_[j] = 0; }
        __syncthreads();

        int j0 = 0;
        for (int guard = 0; guard <= NT + 2; ++guard) {
            if (lane == 0) used_[j0] = 1;
            __syncthreads();
            const int    i0  = p_[j0];
            const double ui0 = u_[i0];

            double best  = INF;
            int    bestj = -1;
            for (int j = 1 + lane; j <= NQ; j += 32) {
                if (!used_[j]) {
                    double cur = (double)costT[i0 - 1][j - 1] - ui0 - v_[j];
                    if (cur < minv[j]) { minv[j] = cur; way_[j] = j0; }
                    if (minv[j] < best) { best = minv[j]; bestj = j; }
                }
            }
            // wave32 argmin (lexicographic: value, then lowest index like np.argmin)
            #pragma unroll
            for (int off = 16; off > 0; off >>= 1) {
                double ov = __shfl_xor(best, off, 32);
                int    oj = __shfl_xor(bestj, off, 32);
                if (ov < best || (ov == best && (unsigned)oj < (unsigned)bestj)) {
                    best = ov; bestj = oj;
                }
            }
            const double delta = best;
            const int    j1    = bestj;
            __syncthreads();

            for (int j = lane; j <= NQ; j += 32) {
                if (used_[j]) { u_[p_[j]] += delta; v_[j] -= delta; }
                else          { minv[j] -= delta; }
            }
            __syncthreads();

            j0 = j1;
            if (j0 < 0 || p_[j0] == 0) break;
        }
        // augmenting path (serial, tiny)
        if (lane == 0) {
            int j = j0;
            while (j != 0) { int jn = way_[j]; p_[j] = p_[jn]; j = jn; }
        }
        __syncthreads();
    }

    // ---- emit assignment + fused matched-pair bbox/giou losses ----
    float bb = 0.f, gg = 0.f;
    for (int j = 1 + lane; j <= NQ; j += 32) {
        int t = p_[j] - 1;
        q2t[b * NQ + (j - 1)] = t;
        if (t >= 0) {
            int s = j - 1;
            const float* pb = pboxes + (size_t)(b * NQ + s) * 4;
            float cx = pb[0], cy = pb[1], w = pb[2], h = pb[3];
            bb += fabsf(cx - traw[t][0]) + fabsf(cy - traw[t][1]) +
                  fabsf(w  - traw[t][2]) + fabsf(h  - traw[t][3]);
            float x1 = cx - 0.5f * w, y1 = cy - 0.5f * h;
            float x2 = cx + 0.5f * w, y2 = cy + 0.5f * h;
            float qarea = (x2 - x1) * (y2 - y1);
            float g = giou_pair(x1, y1, x2, y2, qarea,
                                txy[t][0], txy[t][1], txy[t][2], txy[t][3], tarea[t]);
            gg += 1.0f - g;
        }
    }
    #pragma unroll
    for (int off = 16; off > 0; off >>= 1) {
        bb += __shfl_xor(bb, off, 32);
        gg += __shfl_xor(gg, off, 32);
    }
    if (lane == 0) { bbox_part[b] = bb; giou_part[b] = gg; }
}

// ---------------------------------------------------------------------------
// Kernel 2: sigmoid focal loss over [BS, NQ, NC]; reduction runs on the WMMA
// pipe: acc(16x16 f32) += A(16x4 f32) x ones(4x16). Sum(acc)/16 == sum(terms).
// Target classes materialized once per block into LDS (removes divergent
// dependent gathers from the hot loop); logits fetched as 64-bit pairs.
// ---------------------------------------------------------------------------
__device__ inline float focal_term_x(float x, int c, int tcls)
{
    float t   = (c == tcls) ? 1.f : 0.f;
    float p   = 1.f / (1.f + expf(-x));
    float ce  = fmaxf(x, 0.f) - x * t + log1pf(expf(-fabsf(x)));
    float p_t = p * t + (1.f - p) * (1.f - t);
    float al  = 0.25f * t + 0.75f * (1.f - t);
    float om  = 1.f - p_t;
    return al * ce * om * om;
}

__global__ __launch_bounds__(256)
void focal_kernel(const float* __restrict__ logits,
                  const int*   __restrict__ labels,
                  const int*   __restrict__ q2t,
                  float* __restrict__ ce_part)   // [BS]
{
    const int b    = blockIdx.x;
    const int tid  = threadIdx.x;
    const int wave = tid >> 5;
    const int lane = tid & 31;
    const int total = NQ * NC;                  // 3000

    __builtin_prefetch(logits + (size_t)b * total, 0, 0);   // global_prefetch_b8

    // stage per-query target class once (uniform across the 10 classes / waves)
    __shared__ int tcls_sh[NQ];
    for (int s = tid; s < NQ; s += 256) {
        int tg = q2t[b * NQ + s];
        tcls_sh[s] = (tg >= 0) ? labels[b * NT + tg] : NC;  // NC == "no object"
    }
    __syncthreads();

    const float* lgb = logits + (size_t)b * total;

#ifdef HAVE_WMMA_F32X4
    v8f acc = {0.f, 0.f, 0.f, 0.f, 0.f, 0.f, 0.f, 0.f};
    v2f ones; ones[0] = 1.f; ones[1] = 1.f;
    const int ITER = (total + 8 * 64 - 1) / (8 * 64);       // 6
    for (int k = 0; k < ITER; ++k) {
        int base = (k * 8 + wave) * 64 + lane * 2;          // even, 8B aligned
        v2f a; a[0] = 0.f; a[1] = 0.f;
        if (base + 1 < total) {
            v2f xv = *(const v2f*)(lgb + base);             // global_load_b64
            int s0 = base / NC;
            int c0 = base - s0 * NC;
            int s1 = (c0 == NC - 1) ? s0 + 1 : s0;          // element base+1
            int c1 = (c0 == NC - 1) ? 0 : c0 + 1;
            a[0] = focal_term_x(xv[0], c0, tcls_sh[s0]);
            a[1] = focal_term_x(xv[1], c1, tcls_sh[s1]);
        }
        // v_wmma_f32_16x16x4_f32: acc += A x ones (full-f32 reduction, matrix pipe)
        acc = __builtin_amdgcn_wmma_f32_16x16x4_f32(
                  false, a, false, ones, (short)0, acc, false, false);
    }
    float s = 0.f;
    #pragma unroll
    for (int r = 0; r < 8; ++r) s += acc[r];
#else
    // fallback: f16 WMMA 16x16x32 (codegen-confirmed builtin); 512 terms/wave/op
    v8f acc = {0.f, 0.f, 0.f, 0.f, 0.f, 0.f, 0.f, 0.f};
    v16h ones;
    #pragma unroll
    for (int r = 0; r < 16; ++r) ones[r] = (_Float16)1.0f;
    const int ITER = (total + 8 * 512 - 1) / (8 * 512);     // 1
    for (int k = 0; k < ITER; ++k) {
        int base = (k * 8 + wave) * 512 + lane * 16;
        v16h a;
        #pragma unroll
        for (int r = 0; r < 16; ++r) {
            int e = base + r;
            float term = 0.f;
            if (e < total) {
                int s0 = e / NC;
                int c0 = e - s0 * NC;
                term = focal_term_x(lgb[e], c0, tcls_sh[s0]);
            }
            a[r] = (_Float16)term;
        }
        acc = __builtin_amdgcn_wmma_f32_16x16x32_f16(
                  false, a, false, ones, (short)0, acc, false, false);
    }
    float s = 0.f;
    #pragma unroll
    for (int r = 0; r < 8; ++r) s += acc[r];
#endif
    // each row-sum is replicated across the 16 accumulator columns -> /16 exact
    #pragma unroll
    for (int off = 16; off > 0; off >>= 1) s += __shfl_xor(s, off, 32);
    float wave_sum = s * (1.0f / 16.0f);

    __shared__ float wsum[8];
    if (lane == 0) wsum[wave] = wave_sum;
    __syncthreads();
    if (tid == 0) {
        float tot = 0.f;
        #pragma unroll
        for (int w = 0; w < 8; ++w) tot += wsum[w];
        ce_part[b] = tot;
    }
}

// ---------------------------------------------------------------------------
// Kernel 3: finalize -> scalar loss
// ---------------------------------------------------------------------------
__global__ __launch_bounds__(64)
void finalize_kernel(const float* __restrict__ ce_part,
                     const float* __restrict__ bbox_part,
                     const float* __restrict__ giou_part,
                     float* __restrict__ out)
{
    __shared__ float sh[3][BS];
    int tid = threadIdx.x;
    sh[0][tid] = ce_part[tid];
    sh[1][tid] = bbox_part[tid];
    sh[2][tid] = giou_part[tid];
    __syncthreads();
    if (tid == 0) {
        float ce = 0.f, bb = 0.f, gg = 0.f;
        for (int i = 0; i < BS; ++i) { ce += sh[0][i]; bb += sh[1][i]; gg += sh[2][i]; }
        const float num = (float)(BS * NT);    // 30 matches per batch (n <= m)
        out[0] = ce / num + 5.0f * (bb / num) + 2.0f * (gg / num);
    }
}

// ---------------------------------------------------------------------------
extern "C" void kernel_launch(void* const* d_in, const int* in_sizes, int n_in,
                              void* d_out, int out_size, void* d_ws, size_t ws_size,
                              hipStream_t stream)
{
    const float* pred_logits = (const float*)d_in[0];   // [64,300,10] f32
    const float* pred_boxes  = (const float*)d_in[1];   // [64,300,4]  f32
    const int*   labels      = (const int*)d_in[2];     // [64,30]     int
    const float* boxes       = (const float*)d_in[3];   // [64,30,4]   f32
    float* out = (float*)d_out;

    char* ws = (char*)d_ws;
    int*   q2t       = (int*)ws;                              // 64*300 ints = 76800 B
    float* ce_part   = (float*)(ws + BS * NQ * sizeof(int));  // [64]
    float* bbox_part = ce_part + BS;                          // [64]
    float* giou_part = bbox_part + BS;                        // [64]

    hungarian_kernel<<<BS, 32, 0, stream>>>(pred_logits, pred_boxes, labels, boxes,
                                            q2t, bbox_part, giou_part);
    focal_kernel<<<BS, 256, 0, stream>>>(pred_logits, labels, q2t, ce_part);
    finalize_kernel<<<1, 64, 0, stream>>>(ce_part, bbox_part, giou_part, out);
}
